// Attention_43413529428606
// MI455X (gfx1250) — matrix operand order repaired and checked by reading, not compiled, not run on hardware
//
#include <hip/hip_runtime.h>
#include <hip/hip_bf16.h>

// ---------------------------------------------------------------------------
// Problem constants (from the reference)
// ---------------------------------------------------------------------------
#define HEADS  16
#define HDIM   64
#define DMODEL 1024
#define CTX    8192
#define NB     2

// LDS tile row stride (bf16 elements): 72 halves = 144 B.
//  - 16B-aligned rows => ds_load_b128 still emitted for fragments
//  - 36*row mod 64 banks spreads rows over 16 bank groups (vs 2 at stride 64)
#define LSTR 72

typedef __attribute__((ext_vector_type(16))) __bf16 v16bf;
typedef __attribute__((ext_vector_type(8)))  float  v8f;

// fp32 -> bf16 round-to-nearest-even
__device__ __forceinline__ __bf16 to_bf16(float f) {
    unsigned int u = __builtin_bit_cast(unsigned int, f);
    unsigned int r = u + 0x7FFFu + ((u >> 16) & 1u);
    unsigned short s = (unsigned short)(r >> 16);
    return __builtin_bit_cast(__bf16, s);
}
__device__ __forceinline__ unsigned int pack_bf16(float f0, float f1) {
    unsigned int lo = (unsigned int)__builtin_bit_cast(unsigned short, to_bf16(f0));
    unsigned int hi = (unsigned int)__builtin_bit_cast(unsigned short, to_bf16(f1));
    return lo | (hi << 16);
}

// ---------------------------------------------------------------------------
// WMMA fragment builders (wave32, v_wmma_f32_16x16x32_bf16; contiguous k order, not the ISA table order)
// A (16x32, MxK): lane L: row M=L%16; halves j -> K = 16*(L>>4)+j.
// B (32x16, KxN): lane L: column N=L%16; halves j -> K = 16*(L>>4)+j.
// Offsets 16B-aligned -> ds_load_b128 / global_load_b128.
// ---------------------------------------------------------------------------
__device__ __forceinline__ v16bf frag_a(const __bf16* row0, int stride, int lane) {
    const int m = lane & 15, hi = lane >> 4;
    const __bf16* row = row0 + m * stride;
    v16bf a;
#pragma unroll
    for (int j = 0; j < 16; ++j) a[j] = row[hi * 16 + j];
    return a;
}

__device__ __forceinline__ v16bf frag_b(const __bf16* row0, int stride, int lane) {
    const int n = lane & 15, hi = lane >> 4;
    const __bf16* row = row0 + n * stride + hi * 16;
    v16bf b;
#pragma unroll
    for (int j = 0; j < 16; ++j) b[j] = row[j];
    return b;
}

#define WMMA_BF16(A, B, C) \
    __builtin_amdgcn_wmma_f32_16x16x32_bf16(false, (A), false, (B), (short)0, (C), false, false)

// Stage a [rows x 64] fp32 tile (global, row stride gstride) into bf16 LDS
// (row stride LSTR) as packed b32 stores: conflict-free, 2 elems/thread/step.
__device__ __forceinline__ void stage_tile(const float* __restrict__ g, size_t gstride,
                                           __bf16* s, int rows, int tid) {
    for (int p = tid; p < rows * 32; p += 256) {
        const int r = p >> 5, c2 = (p & 31) * 2;
        const float f0 = g[(size_t)r * gstride + c2];
        const float f1 = g[(size_t)r * gstride + c2 + 1];
        *(unsigned int*)&s[r * LSTR + c2] = pack_bf16(f0, f1);
    }
}

// ---------------------------------------------------------------------------
// Kernel 0: zero the kv accumulation buffer (d_ws is poisoned by the harness)
// ---------------------------------------------------------------------------
__global__ void zero_f32_kernel(float* __restrict__ p, int n) {
    int i = blockIdx.x * blockDim.x + threadIdx.x;
    if (i < n) p[i] = 0.0f;
}

// ---------------------------------------------------------------------------
// Kernel 1: fused Q/K/V projection + epilogue.
//   grid = (CTX/64, DMODEL/256, 3*NB); block = 256 (8 wave32)
//   Block: O[64 tokens x 256 feats (4 heads)] = X[64x1024] @ W[256x1024]^T
//   Wave tile 2x4; all 6 fragments preloaded per K=32 step (partial dscnt waits).
//   sC (f32 epilogue buffer, stride 257) aliases sA/sB - dead after K loop.
// ---------------------------------------------------------------------------
__global__ __launch_bounds__(256) void qkv_proj_kernel(
    const float* __restrict__ x, const int* __restrict__ mask,
    const float* __restrict__ Wq, const float* __restrict__ Wk,
    const float* __restrict__ Wv, const float* __restrict__ ncst,
    unsigned short* __restrict__ qkv_raw)
{
    __shared__ __align__(16) char smem[64 * 257 * 4];      // 65792 B
    __bf16* sA = (__bf16*)smem;                            // [64 ][LSTR]  9216 B
    __bf16* sB = (__bf16*)(smem + 64 * LSTR * 2);          // [256][LSTR] 36864 B
    float*  sC = (float*)smem;                             // [64][257] (aliases)

    __bf16* qkv = (__bf16*)qkv_raw;
    const int tid = threadIdx.x, lane = tid & 31, wave = tid >> 5;
    const int c0  = blockIdx.x * 64;
    const int o0  = blockIdx.y * 256;          // first output feature
    const int mat = blockIdx.z / NB, b = blockIdx.z % NB;
    const float* W = (mat == 0) ? Wq : (mat == 1) ? Wk : Wv;

    const int tm0 = (wave & 1) * 2;            // 2 M tiles
    const int tn0 = (wave >> 1) * 4;           // 4 N tiles
    v8f acc[2][4] = {};

    const size_t xrow = ((size_t)b * CTX + c0) * DMODEL;

    for (int kt = 0; kt < DMODEL / 64; ++kt) {
        const int k0 = kt * 64;
        stage_tile(x + xrow + k0, DMODEL, sA, 64, tid);
        stage_tile(W + (size_t)o0 * DMODEL + k0, DMODEL, sB, 256, tid);
        if (kt + 1 < DMODEL / 64) {                       // gfx1250 global_prefetch_b8
            __builtin_prefetch(&x[xrow + (size_t)(tid >> 2) * DMODEL + k0 + 64], 0, 1);
            __builtin_prefetch(&W[(size_t)(o0 + (tid >> 1)) * DMODEL + k0 + 64], 0, 1);
        }
        __syncthreads();
#pragma unroll
        for (int ks = 0; ks < 64; ks += 32) {
            v16bf a0 = frag_a(sA + (tm0 + 0) * 16 * LSTR + ks, LSTR, lane);
            v16bf a1 = frag_a(sA + (tm0 + 1) * 16 * LSTR + ks, LSTR, lane);
            v16bf b0 = frag_b(sB + (tn0 + 0) * 16 * LSTR + ks, LSTR, lane);
            v16bf b1 = frag_b(sB + (tn0 + 1) * 16 * LSTR + ks, LSTR, lane);
            v16bf b2 = frag_b(sB + (tn0 + 2) * 16 * LSTR + ks, LSTR, lane);
            v16bf b3 = frag_b(sB + (tn0 + 3) * 16 * LSTR + ks, LSTR, lane);
            acc[0][0] = WMMA_BF16(a0, b0, acc[0][0]);
            acc[1][0] = WMMA_BF16(a1, b0, acc[1][0]);
            acc[0][1] = WMMA_BF16(a0, b1, acc[0][1]);
            acc[1][1] = WMMA_BF16(a1, b1, acc[1][1]);
            acc[0][2] = WMMA_BF16(a0, b2, acc[0][2]);
            acc[1][2] = WMMA_BF16(a1, b2, acc[1][2]);
            acc[0][3] = WMMA_BF16(a0, b3, acc[0][3]);
            acc[1][3] = WMMA_BF16(a1, b3, acc[1][3]);
        }
        __syncthreads();
    }

    // spill accumulators to LDS (stride 257 f32 -> conflict-free epilogue)
    {
        const int n = lane & 15, hi = lane >> 4;
#pragma unroll
        for (int i = 0; i < 2; ++i)
#pragma unroll
            for (int j = 0; j < 4; ++j)
#pragma unroll
                for (int r = 0; r < 8; ++r) {
                    const int row = (tm0 + i) * 16 + hi * 8 + r;
                    sC[row * 257 + (tn0 + j) * 16 + n] = acc[i][j][r];
                }
    }
    __syncthreads();

    // epilogue: thread t -> token row (t&63), head slot (t>>6)
    {
        const int m = tid & 63, q = tid >> 6;
        const int c = c0 + m;
        const int h = (o0 >> 6) + q;
        const float* row = sC + m * 257 + q * 64;
        __bf16* dst = qkv + ((((size_t)mat * NB + b) * HEADS + h) * CTX + c) * HDIM;
        if (mat < 2) {  // q or k: l2norm + interleaved RoPE + mask
            float ss = 0.0f;
#pragma unroll
            for (int j = 0; j < 64; ++j) ss += row[j] * row[j];
            const float inv = 1.0f / fmaxf(sqrtf(ss), 1e-12f);
            const float mk  = (float)mask[(size_t)b * CTX + c];
            const float lf  = -9.210340371976184f / 32.0f;  // -ln(10000)/32
#pragma unroll
            for (int i = 0; i < 32; ++i) {
                const float fr  = __expf(lf * (float)i);    // theta^{-2i/64}
                const float ang = (float)c * fr;
                float s, cs;
                __sincosf(ang, &s, &cs);
                const float x0 = row[2 * i] * inv, x1 = row[2 * i + 1] * inv;
                dst[2 * i]     = to_bf16((x0 * cs - x1 * s) * mk);
                dst[2 * i + 1] = to_bf16((x1 * cs + x0 * s) * mk);
            }
        } else {        // v: divide by C ** sigmoid(norm_const[h])
            const float sig   = 1.0f / (1.0f + __expf(-ncst[h]));
            const float scale = __expf(-sig * 9.010913347279288f);  // ln(8192)
#pragma unroll
            for (int j = 0; j < 64; ++j) dst[j] = to_bf16(row[j] * scale);
        }
    }
}

// ---------------------------------------------------------------------------
// Kernel 2: kv[b,h] = k^T @ v  (64x64, K=CTX split 8 ways, f32 atomic merge).
// k,v staged TRANSPOSED ([d][c], stride LSTR, packed b32 writes along c).
//   grid = (8, HEADS, NB); block = 256
// ---------------------------------------------------------------------------
__global__ __launch_bounds__(256) void kv_kernel(
    const unsigned short* __restrict__ qkv_raw, float* __restrict__ kv)
{
    __shared__ __bf16 sKT[64 * LSTR];   // k^T tile: [d][c]
    __shared__ __bf16 sVT[64 * LSTR];   // v^T tile: [e][c]

    const __bf16* qkv = (const __bf16*)qkv_raw;
    const int tid = threadIdx.x, lane = tid & 31, wave = tid >> 5;
    const int chunk = blockIdx.x, h = blockIdx.y, b = blockIdx.z;
    const int tm = wave >> 1, tn0 = (wave & 1) * 2;

    const __bf16* kbase = qkv + (((size_t)1 * NB + b) * HEADS + h) * (size_t)CTX * HDIM;
    const __bf16* vbase = qkv + (((size_t)2 * NB + b) * HEADS + h) * (size_t)CTX * HDIM;

    v8f acc0 = {}, acc1 = {};
    for (int it = 0; it < 16; ++it) {
        const int cb = chunk * 1024 + it * 64;
        // transpose-stage: pack two consecutive c into one b32 LDS store
#pragma unroll 4
        for (int p = tid; p < 2048; p += 256) {
            const int d = p & 63, rp = (p >> 6) * 2;     // d: dim, rp: even c-offset
            *(unsigned int*)&sKT[d * LSTR + rp] = pack_bf16(
                (float)kbase[(size_t)(cb + rp) * HDIM + d],
                (float)kbase[(size_t)(cb + rp + 1) * HDIM + d]);
            *(unsigned int*)&sVT[d * LSTR + rp] = pack_bf16(
                (float)vbase[(size_t)(cb + rp) * HDIM + d],
                (float)vbase[(size_t)(cb + rp + 1) * HDIM + d]);
        }
        __syncthreads();
#pragma unroll
        for (int ks = 0; ks < 64; ks += 32) {
            v16bf a  = frag_a(sKT + tm * 16 * LSTR + ks, LSTR, lane);
            v16bf b0 = frag_b(sVT + tn0 * 16 * LSTR + ks, LSTR, lane);
            v16bf b1 = frag_b(sVT + (tn0 + 1) * 16 * LSTR + ks, LSTR, lane);
            acc0 = WMMA_BF16(a, b0, acc0);
            acc1 = WMMA_BF16(a, b1, acc1);
        }
        __syncthreads();
    }

    float* out = kv + ((size_t)b * HEADS + h) * HDIM * HDIM;
    const int n = lane & 15, hi = lane >> 4;
#pragma unroll
    for (int r = 0; r < 8; ++r) {
        const int row = tm * 16 + hi * 8 + r;
        __hip_atomic_fetch_add(&out[row * 64 + tn0 * 16 + n], acc0[r],
                               __ATOMIC_RELAXED, __HIP_MEMORY_SCOPE_AGENT);
        __hip_atomic_fetch_add(&out[row * 64 + (tn0 + 1) * 16 + n], acc1[r],
                               __ATOMIC_RELAXED, __HIP_MEMORY_SCOPE_AGENT);
    }
}

// ---------------------------------------------------------------------------
// Kernel 3: attn = q @ kv  (M=CTX, N=64/head, K=64). A fragments straight from
// global bf16 q; kv transposed into bf16 LDS. 128-token blocks, 2x2 tiles/wave.
//   grid = (CTX/128, HEADS, NB); block = 256
// ---------------------------------------------------------------------------
__global__ __launch_bounds__(256) void attn_kernel(
    const unsigned short* __restrict__ qkv_raw, const float* __restrict__ kv,
    unsigned short* __restrict__ attn_raw)
{
    __shared__ __bf16 sKVT[64 * LSTR];   // kv^T as bf16: [e][d]

    const __bf16* qkv = (const __bf16*)qkv_raw;
    __bf16* attn = (__bf16*)attn_raw;
    const int tid = threadIdx.x, lane = tid & 31, wave = tid >> 5;
    const int c0 = blockIdx.x * 128, h = blockIdx.y, b = blockIdx.z;
    const int tm0 = (wave >> 1) * 2;          // 2 of 8 M tiles
    const int tn0 = (wave & 1) * 2;           // 2 of 4 N tiles

    const float* kvp = kv + ((size_t)b * HEADS + h) * HDIM * HDIM;
#pragma unroll
    for (int p = tid; p < 2048; p += 256) {
        const int e = p >> 5, d2 = (p & 31) * 2;         // transpose + pack pairs
        *(unsigned int*)&sKVT[e * LSTR + d2] =
            pack_bf16(kvp[d2 * 64 + e], kvp[(d2 + 1) * 64 + e]);
    }
    const __bf16* qbase = qkv + (((size_t)0 * NB + b) * HEADS + h) * (size_t)CTX * HDIM;
    __syncthreads();

    v8f acc[2][2] = {};
#pragma unroll
    for (int ks = 0; ks < 64; ks += 32) {
        v16bf a0 = frag_a(qbase + (size_t)(c0 + (tm0 + 0) * 16) * HDIM + ks, HDIM, lane);
        v16bf a1 = frag_a(qbase + (size_t)(c0 + (tm0 + 1) * 16) * HDIM + ks, HDIM, lane);
        v16bf b0 = frag_b(sKVT + (tn0 + 0) * 16 * LSTR + ks, LSTR, lane);
        v16bf b1 = frag_b(sKVT + (tn0 + 1) * 16 * LSTR + ks, LSTR, lane);
        acc[0][0] = WMMA_BF16(a0, b0, acc[0][0]);
        acc[1][0] = WMMA_BF16(a1, b0, acc[1][0]);
        acc[0][1] = WMMA_BF16(a0, b1, acc[0][1]);
        acc[1][1] = WMMA_BF16(a1, b1, acc[1][1]);
    }

    const int n = lane & 15, hi = lane >> 4;
#pragma unroll
    for (int i = 0; i < 2; ++i)
#pragma unroll
        for (int r = 0; r < 8; ++r) {
            const int c = c0 + (tm0 + i) * 16 + hi * 8 + r;
            __bf16* dst = attn + ((size_t)b * CTX + c) * DMODEL + h * HDIM;
            dst[tn0 * 16 + n]       = to_bf16(acc[i][0][r]);
            dst[(tn0 + 1) * 16 + n] = to_bf16(acc[i][1][r]);
        }
}

// ---------------------------------------------------------------------------
// Kernel 4: out = attn @ Wo^T  ([NB*CTX x 1024] x [1024 x 1024]), fp32 out.
//   grid = (NB*CTX/64, DMODEL/256); block = 256; 2x4 tiles per wave.
//   A fragments loaded straight from global bf16 (K-contiguous rows).
// ---------------------------------------------------------------------------
__global__ __launch_bounds__(256) void out_proj_kernel(
    const unsigned short* __restrict__ attn_raw, const float* __restrict__ Wo,
    float* __restrict__ out)
{
    __shared__ __bf16 sB[256 * LSTR];   // Wo tile [out_feat][k]

    const __bf16* attn = (const __bf16*)attn_raw;
    const int tid = threadIdx.x, lane = tid & 31, wave = tid >> 5;
    const int t0 = blockIdx.x * 64;    // flat token tile over NB*CTX
    const int o0 = blockIdx.y * 256;   // output-feature tile
    const int tm0 = (wave & 1) * 2;
    const int tn0 = (wave >> 1) * 4;

    v8f acc[2][4] = {};
    for (int kt = 0; kt < DMODEL / 64; ++kt) {
        const int k0 = kt * 64;
        stage_tile(Wo + (size_t)o0 * DMODEL + k0, DMODEL, sB, 256, tid);
        if (kt + 1 < DMODEL / 64)
            __builtin_prefetch(&Wo[(size_t)(o0 + (tid >> 1)) * DMODEL + k0 + 64], 0, 1);
        __syncthreads();
#pragma unroll
        for (int ks = 0; ks < 64; ks += 32) {
            v16bf a0 = frag_a(attn + (size_t)(t0 + (tm0 + 0) * 16) * DMODEL + k0 + ks, DMODEL, lane);
            v16bf a1 = frag_a(attn + (size_t)(t0 + (tm0 + 1) * 16) * DMODEL + k0 + ks, DMODEL, lane);
            v16bf b0 = frag_b(sB + (tn0 + 0) * 16 * LSTR + ks, LSTR, lane);
            v16bf b1 = frag_b(sB + (tn0 + 1) * 16 * LSTR + ks, LSTR, lane);
            v16bf b2 = frag_b(sB + (tn0 + 2) * 16 * LSTR + ks, LSTR, lane);
            v16bf b3 = frag_b(sB + (tn0 + 3) * 16 * LSTR + ks, LSTR, lane);
            acc[0][0] = WMMA_BF16(a0, b0, acc[0][0]);
            acc[1][0] = WMMA_BF16(a1, b0, acc[1][0]);
            acc[0][1] = WMMA_BF16(a0, b1, acc[0][1]);
            acc[1][1] = WMMA_BF16(a1, b1, acc[1][1]);
            acc[0][2] = WMMA_BF16(a0, b2, acc[0][2]);
            acc[1][2] = WMMA_BF16(a1, b2, acc[1][2]);
            acc[0][3] = WMMA_BF16(a0, b3, acc[0][3]);
            acc[1][3] = WMMA_BF16(a1, b3, acc[1][3]);
        }
        __syncthreads();
    }

    const int n = lane & 15, hi = lane >> 4;
#pragma unroll
    for (int i = 0; i < 2; ++i)
#pragma unroll
        for (int j = 0; j < 4; ++j)
#pragma unroll
            for (int r = 0; r < 8; ++r) {
                const int t = t0 + (tm0 + i) * 16 + hi * 8 + r;
                out[(size_t)t * DMODEL + o0 + (tn0 + j) * 16 + n] = acc[i][j][r];
            }
}

// ---------------------------------------------------------------------------
// Launch
// ---------------------------------------------------------------------------
extern "C" void kernel_launch(void* const* d_in, const int* in_sizes, int n_in,
                              void* d_out, int out_size, void* d_ws, size_t ws_size,
                              hipStream_t stream) {
    const float* x    = (const float*)d_in[0];
    const int*   mask = (const int*)d_in[1];
    const float* Wq   = (const float*)d_in[2];
    const float* Wk   = (const float*)d_in[3];
    const float* Wv   = (const float*)d_in[4];
    const float* Wo   = (const float*)d_in[5];
    const float* nc   = (const float*)d_in[6];
    float* out = (float*)d_out;

    // workspace layout (bytes):
    //   [0, qkv_bytes)            : q,k,v bf16  [3][NB][H][CTX][HD]   (96 MiB)
    //   [qkv_bytes, +attn_bytes)  : attn bf16   [NB][CTX][DMODEL]     (32 MiB)
    //   [.., +kv_bytes)           : kv f32      [NB][H][HD][HD]       (0.5 MiB)
    char* ws = (char*)d_ws;
    const size_t qkv_bytes  = (size_t)3 * NB * HEADS * CTX * HDIM * 2;
    const size_t attn_bytes = (size_t)NB * CTX * DMODEL * 2;
    unsigned short* qkv  = (unsigned short*)ws;
    unsigned short* attn = (unsigned short*)(ws + qkv_bytes);
    float*          kvb  = (float*)(ws + qkv_bytes + attn_bytes);
    const int kv_elems = NB * HEADS * HDIM * HDIM;   // 131072

    zero_f32_kernel<<<dim3((kv_elems + 255) / 256), dim3(256), 0, stream>>>(kvb, kv_elems);

    qkv_proj_kernel<<<dim3(CTX / 64, DMODEL / 256, 3 * NB), dim3(256), 0, stream>>>(
        x, mask, Wq, Wk, Wv, nc, qkv);

    kv_kernel<<<dim3(8, HEADS, NB), dim3(256), 0, stream>>>(qkv, kvb);

    attn_kernel<<<dim3(CTX / 128, HEADS, NB), dim3(256), 0, stream>>>(qkv, kvb, attn);

    out_proj_kernel<<<dim3(NB * CTX / 64, DMODEL / 256), dim3(256), 0, stream>>>(attn, Wo, out);
}
